// SlotAttention_20272245637247
// MI455X (gfx1250) — compile-verified, matrix operand order checked by reading, not run on hardware
//
#include <hip/hip_runtime.h>
#include <hip/hip_bf16.h>

// ---------------- problem sizes (fixed by reference) ----------------
#define BB   64
#define NN   4096
#define DD   64
#define SS   8
#define HH   128
#define NITR 3

typedef __attribute__((ext_vector_type(16))) __bf16 v16bf;
typedef __attribute__((ext_vector_type(8)))  __bf16 bf16x8;
typedef __attribute__((ext_vector_type(8)))  float  v8f;

static __device__ __forceinline__ v8f wmma_bf16(v16bf a, v16bf b, v8f c) {
    return __builtin_amdgcn_wmma_f32_16x16x32_bf16(false, a, false, b, (short)0, c,
                                                   false, false);
}

// Load two contiguous 8-element bf16 groups (16B each) and splice into a fragment.
static __device__ __forceinline__ v16bf ldfrag(const __bf16* lo, const __bf16* hi) {
    bf16x8 l = *(const bf16x8*)lo;
    bf16x8 h = *(const bf16x8*)hi;
    return __builtin_shufflevector(l, h, 0, 1, 2, 3, 4, 5, 6, 7,
                                          8, 9, 10, 11, 12, 13, 14, 15);
}

// -------------------------------------------------------------------
// slots = mu + exp(log_sigma) * noise
// -------------------------------------------------------------------
__global__ void k_init_slots(const float* __restrict__ mu, const float* __restrict__ ls,
                             const float* __restrict__ noise, float* __restrict__ slots) {
    int i = blockIdx.x * 256 + threadIdx.x;
    if (i < BB * SS * DD) {
        int d = i & (DD - 1);
        slots[i] = mu[d] + __expf(ls[d]) * noise[i];
    }
}

// -------------------------------------------------------------------
// One-time zeroing of attn padding rows 8..15 (A-matrix padding for attn@v).
// attn is [b][16][NN] bf16; rows 8..15 stay zero across all iterations.
// -------------------------------------------------------------------
__global__ void k_zero_attn_pad(__bf16* __restrict__ attn) {
    // total pad elements: BB * 8 * NN; write 8 bf16 per thread
    const size_t idx = ((size_t)blockIdx.x * 256 + threadIdx.x) * 8;
    const size_t perB = (size_t)SS * NN;            // pad elems per batch
    const size_t b = idx / perB;
    const size_t r = idx % perB;                    // within rows 8..15 block
    bf16x8 z = {};
    *(bf16x8*)(attn + b * 16 * NN + (size_t)SS * NN + r) = z;
}

// -------------------------------------------------------------------
// Fused LN(inputs) -> k = x@Wk+bk (bf16 [b][n][d]) and v -> vT (bf16 [b][d][n])
// 256 threads = 8 waves, each wave owns a 16-row tile; grid = B*N/128
// Weights staged TRANSPOSED in LDS so B-fragment K-runs are contiguous.
// -------------------------------------------------------------------
__launch_bounds__(256)
__global__ void k_lnkv(const float* __restrict__ inp,
                       const float* __restrict__ g_in, const float* __restrict__ be_in,
                       const float* __restrict__ wkf, const float* __restrict__ bkf,
                       const float* __restrict__ wvf, const float* __restrict__ bvf,
                       __bf16* __restrict__ kmat, __bf16* __restrict__ vT) {
    __shared__ alignas(16) __bf16 sWkT[DD * DD];     // [n][k]
    __shared__ alignas(16) __bf16 sWvT[DD * DD];     // [n][k]
    __shared__ alignas(16) __bf16 sX[8][16 * DD];

    const int t = threadIdx.x;
    for (int i = t; i < DD * DD; i += 256) {
        const int k = i >> 6, n = i & 63;
        sWkT[n * DD + k] = (__bf16)wkf[i];
        sWvT[n * DD + k] = (__bf16)wvf[i];
    }
    __syncthreads();

    const int wave = t >> 5, lane = t & 31;
    const int row16 = lane & 15, hlf = lane >> 4;
    const int b = blockIdx.x >> 5;               // 32 blocks per batch
    const int nbase = (blockIdx.x & 31) * 128 + wave * 16;

    // LayerNorm 16 rows: each lane handles 2 consecutive columns, wave-reduce stats
    const float g0 = g_in[lane * 2], g1 = g_in[lane * 2 + 1];
    const float e0 = be_in[lane * 2], e1 = be_in[lane * 2 + 1];
    for (int r = 0; r < 16; ++r) {
        const float* p = inp + ((size_t)(b * NN + nbase + r)) * DD + lane * 2;
        float x0 = p[0], x1 = p[1];
        float s = x0 + x1, q = x0 * x0 + x1 * x1;
        for (int off = 16; off > 0; off >>= 1) {
            s += __shfl_xor(s, off, 32);
            q += __shfl_xor(q, off, 32);
        }
        float m  = s * (1.0f / 64.0f);
        float vr = q * (1.0f / 64.0f) - m * m;
        float rs = rsqrtf(vr + 1e-5f);
        sX[wave][r * DD + lane * 2]     = (__bf16)((x0 - m) * rs * g0 + e0);
        sX[wave][r * DD + lane * 2 + 1] = (__bf16)((x1 - m) * rs * g1 + e1);
    }
    __syncthreads();

    // A fragments (16x32 bf16, K blocks 0..31 and 32..63): 4 x b128 LDS loads
    const __bf16* xr = &sX[wave][row16 * DD];
    const v16bf a0 = ldfrag(xr + hlf * 8,      xr + 16 + hlf * 8);
    const v16bf a1 = ldfrag(xr + 32 + hlf * 8, xr + 48 + hlf * 8);

    for (int nb = 0; nb < 4; ++nb) {
        const int nc = nb * 16 + row16;
        const __bf16* wkr = sWkT + nc * DD;
        const __bf16* wvr = sWvT + nc * DD;
        const v16bf bk0 = ldfrag(wkr + hlf * 8,      wkr + 16 + hlf * 8);
        const v16bf bk1 = ldfrag(wkr + 32 + hlf * 8, wkr + 48 + hlf * 8);
        const v16bf bv0 = ldfrag(wvr + hlf * 8,      wvr + 16 + hlf * 8);
        const v16bf bv1 = ldfrag(wvr + 32 + hlf * 8, wvr + 48 + hlf * 8);

        v8f ck = {}, cv = {};
        ck = wmma_bf16(a0, bk0, ck);
        ck = wmma_bf16(a1, bk1, ck);
        cv = wmma_bf16(a0, bv0, cv);
        cv = wmma_bf16(a1, bv1, cv);

        const float bkc = bkf[nc], bvc = bvf[nc];
        // k store: coalesced across lanes (16 x bf16 per row j)
#pragma unroll
        for (int j = 0; j < 8; ++j) {
            const int rown = nbase + j + 8 * hlf;
            kmat[((size_t)(b * NN) + rown) * DD + nc] = (__bf16)(ck[j] + bkc);
        }
        // vT store: a lane's 8 rows are 8 consecutive n -> one 16B store
        bf16x8 pk;
#pragma unroll
        for (int j = 0; j < 8; ++j) pk[j] = (__bf16)(cv[j] + bvc);
        *(bf16x8*)(vT + ((size_t)b * DD + nc) * NN + nbase + 8 * hlf) = pk;
    }
}

// -------------------------------------------------------------------
// q = LN(slots) @ Wq + bq  -> bf16 [b][16][64] (rows 8..15 zero)
// also zeroes rowsum[b][*] and upd[b][*].  64 threads per batch.
// -------------------------------------------------------------------
__launch_bounds__(64)
__global__ void k_qproj(const float* __restrict__ slots,
                        const float* __restrict__ g_sl, const float* __restrict__ be_sl,
                        const float* __restrict__ wq, const float* __restrict__ bq,
                        __bf16* __restrict__ qmat, float* __restrict__ rowsum,
                        float* __restrict__ upd) {
    const int b = blockIdx.x, t = threadIdx.x;
    __shared__ float raw[SS * DD];
    __shared__ float xn[SS * DD];
    for (int s = 0; s < SS; ++s) raw[s * DD + t] = slots[((size_t)b * SS + s) * DD + t];
    __syncthreads();
    for (int s = 0; s < SS; ++s) {
        float m = 0.f, q = 0.f;
        for (int i = 0; i < DD; ++i) { float v = raw[s * DD + i]; m += v; q += v * v; }
        m *= (1.0f / 64.0f);
        float vr = q * (1.0f / 64.0f) - m * m;
        xn[s * DD + t] = (raw[s * DD + t] - m) * rsqrtf(vr + 1e-5f) * g_sl[t] + be_sl[t];
    }
    __syncthreads();
    const float bqt = bq[t];
    for (int s = 0; s < SS; ++s) {
        float acc = bqt;
        for (int i = 0; i < DD; ++i) acc = fmaf(xn[s * DD + i], wq[i * DD + t], acc);
        qmat[((size_t)b * 16 + s) * DD + t] = (__bf16)acc;
    }
    for (int s = SS; s < 16; ++s) qmat[((size_t)b * 16 + s) * DD + t] = (__bf16)0.0f;
    if (t < SS) rowsum[b * SS + t] = 0.f;
    for (int i = t; i < SS * DD; i += 64) upd[(size_t)b * SS * DD + i] = 0.f;
}

// -------------------------------------------------------------------
// dots = q@k^T * scale, softmax over slots (fits in one lane), +eps,
// store attn bf16 [b][s][n] (only s<8; pad rows pre-zeroed once);
// rowsum via LDS + 1 atomic per slot per block.
// 8 waves/block, wave = 16 columns of n; grid = B*32.
// -------------------------------------------------------------------
__launch_bounds__(256)
__global__ void k_dots(const __bf16* __restrict__ qmat, const __bf16* __restrict__ kmat,
                       __bf16* __restrict__ attn, float* __restrict__ rowsum) {
    __shared__ float sRow[SS];
    const int t = threadIdx.x, lane = t & 31;
    const int row16 = lane & 15, hlf = lane >> 4;
    const int b = blockIdx.x >> 5;
    const int n = (blockIdx.x & 31) * 128 + (t >> 5) * 16 + row16;

    if (t < SS) sRow[t] = 0.f;
    __syncthreads();

    const __bf16* qr = qmat + ((size_t)b * 16 + row16) * DD;
    const __bf16* kr = kmat + ((size_t)(b * NN) + n) * DD;
    const v16bf a0 = ldfrag(qr + hlf * 8,      qr + 16 + hlf * 8);
    const v16bf a1 = ldfrag(qr + 32 + hlf * 8, qr + 48 + hlf * 8);
    const v16bf k0 = ldfrag(kr + hlf * 8,      kr + 16 + hlf * 8);
    const v16bf k1 = ldfrag(kr + 32 + hlf * 8, kr + 48 + hlf * 8);

    v8f c = {};
    c = wmma_bf16(a0, k0, c);
    c = wmma_bf16(a1, k1, c);

    // lanes 0..15: accumulators c[0..7] hold dots for slots 0..7 of column n
    float p[8];
    if (hlf == 0) {
        float mx = -3.0e38f;
#pragma unroll
        for (int j = 0; j < 8; ++j) { p[j] = c[j] * 0.125f; mx = fmaxf(mx, p[j]); }
        float sum = 0.f;
#pragma unroll
        for (int j = 0; j < 8; ++j) { p[j] = __expf(p[j] - mx); sum += p[j]; }
        float inv = 1.0f / sum;
#pragma unroll
        for (int j = 0; j < 8; ++j) p[j] = p[j] * inv + 1e-8f;

        __bf16* ap = attn + (size_t)b * 16 * NN;
#pragma unroll
        for (int j = 0; j < 8; ++j) ap[(size_t)j * NN + n] = (__bf16)p[j];
    } else {
#pragma unroll
        for (int j = 0; j < 8; ++j) p[j] = 0.f;
    }

    // rowsum: wave-reduce, accumulate in LDS, single global atomic per slot/block
#pragma unroll
    for (int j = 0; j < 8; ++j) {
        float v = p[j];
        for (int off = 16; off > 0; off >>= 1) v += __shfl_xor(v, off, 32);
        if (lane == 0) atomicAdd(&sRow[j], v);
    }
    __syncthreads();
    if (t < SS) atomicAdd(&rowsum[b * SS + t], sRow[t]);
}

// -------------------------------------------------------------------
// upd += attn @ v  (split-K over 4 N-chunks; normalization deferred to GRU)
// grid = B*4 (b, chunk); 4 waves/block, wave = one 16-wide d tile.
// -------------------------------------------------------------------
__launch_bounds__(128)
__global__ void k_updates(const __bf16* __restrict__ attn, const __bf16* __restrict__ vT,
                          float* __restrict__ upd) {
    const int t = threadIdx.x, wave = t >> 5, lane = t & 31;
    const int row16 = lane & 15, hlf = lane >> 4;
    const int b = blockIdx.x >> 2;
    const int kbeg = (blockIdx.x & 3) * (NN / 4);
    const int dcol = wave * 16 + row16;

    const __bf16* ar = attn + ((size_t)b * 16 + row16) * NN;
    const __bf16* vr = vT + ((size_t)b * DD + dcol) * NN;

    v8f c = {};
    for (int k0 = kbeg; k0 < kbeg + NN / 4; k0 += 32) {
        __builtin_prefetch(ar + k0 + 256, 0, 1);   // one chunk ahead
        __builtin_prefetch(vr + k0 + 256, 0, 1);
        const v16bf a  = ldfrag(ar + k0 + hlf * 8, ar + k0 + 16 + hlf * 8);
        const v16bf bb = ldfrag(vr + k0 + hlf * 8, vr + k0 + 16 + hlf * 8);
        c = wmma_bf16(a, bb, c);
    }
    if (hlf == 0) {
#pragma unroll
        for (int j = 0; j < 8; ++j)
            atomicAdd(&upd[((size_t)b * SS + j) * DD + dcol], c[j]);
    }
}

// -------------------------------------------------------------------
// GRUCell + residual MLP for one (b,s); includes the deferred /rowsum.
// 64 threads per block; grid = B*S.
// -------------------------------------------------------------------
__launch_bounds__(64)
__global__ void k_gru_mlp(const float* __restrict__ upd, const float* __restrict__ rowsum,
                          const float* __restrict__ hprev_g,
                          const float* __restrict__ w_ih, const float* __restrict__ b_ih,
                          const float* __restrict__ w_hh, const float* __restrict__ b_hh,
                          const float* __restrict__ g_ff, const float* __restrict__ be_ff,
                          const float* __restrict__ w1, const float* __restrict__ b1,
                          const float* __restrict__ w2, const float* __restrict__ b2,
                          float* __restrict__ slots_out) {
    const int bs = blockIdx.x, t = threadIdx.x;
    __shared__ float u[DD], hp[DD], gx[3 * DD], gh[3 * DD], h[DD], ff[DD], hid[HH];

    u[t]  = upd[(size_t)bs * DD + t] / rowsum[bs];   // deferred attn renormalization
    hp[t] = hprev_g[(size_t)bs * DD + t];
    __syncthreads();

    for (int j = t; j < 3 * DD; j += 64) {
        float ax = b_ih[j], ah = b_hh[j];
        for (int i = 0; i < DD; ++i) {
            ax = fmaf(u[i],  w_ih[i * 3 * DD + j], ax);
            ah = fmaf(hp[i], w_hh[i * 3 * DD + j], ah);
        }
        gx[j] = ax; gh[j] = ah;
    }
    __syncthreads();
    {
        float r  = 1.0f / (1.0f + __expf(-(gx[t] + gh[t])));
        float z  = 1.0f / (1.0f + __expf(-(gx[DD + t] + gh[DD + t])));
        float nw = tanhf(gx[2 * DD + t] + r * gh[2 * DD + t]);
        h[t] = (1.0f - z) * nw + z * hp[t];
    }
    __syncthreads();
    float m = 0.f, q = 0.f;
    for (int i = 0; i < DD; ++i) { float v = h[i]; m += v; q += v * v; }
    m *= (1.0f / 64.0f);
    float vr = q * (1.0f / 64.0f) - m * m;
    ff[t] = (h[t] - m) * rsqrtf(vr + 1e-5f) * g_ff[t] + be_ff[t];
    __syncthreads();
    for (int j = t; j < HH; j += 64) {
        float acc = b1[j];
        for (int i = 0; i < DD; ++i) acc = fmaf(ff[i], w1[i * HH + j], acc);
        hid[j] = fmaxf(acc, 0.0f);
    }
    __syncthreads();
    {
        float acc = b2[t];
        for (int i = 0; i < HH; ++i) acc = fmaf(hid[i], w2[i * DD + t], acc);
        slots_out[(size_t)bs * DD + t] = h[t] + acc;
    }
}

// -------------------------------------------------------------------
extern "C" void kernel_launch(void* const* d_in, const int* in_sizes, int n_in,
                              void* d_out, int out_size, void* d_ws, size_t ws_size,
                              hipStream_t stream) {
    const float* inputs  = (const float*)d_in[0];
    const float* noise   = (const float*)d_in[1];
    const float* mu      = (const float*)d_in[2];
    const float* lsig    = (const float*)d_in[3];
    const float* wq      = (const float*)d_in[4];
    const float* bq      = (const float*)d_in[5];
    const float* wk      = (const float*)d_in[6];
    const float* bk      = (const float*)d_in[7];
    const float* wv      = (const float*)d_in[8];
    const float* bv      = (const float*)d_in[9];
    const float* w_ih    = (const float*)d_in[10];
    const float* b_ih    = (const float*)d_in[11];
    const float* w_hh    = (const float*)d_in[12];
    const float* b_hh    = (const float*)d_in[13];
    const float* w1      = (const float*)d_in[14];
    const float* b1      = (const float*)d_in[15];
    const float* w2      = (const float*)d_in[16];
    const float* b2      = (const float*)d_in[17];
    const float* g_in    = (const float*)d_in[18];
    const float* beta_in = (const float*)d_in[19];
    const float* g_sl    = (const float*)d_in[20];
    const float* beta_sl = (const float*)d_in[21];
    const float* g_ff    = (const float*)d_in[22];
    const float* beta_ff = (const float*)d_in[23];

    // ---- workspace carve-up ----
    char* ws = (char*)d_ws;
    size_t off = 0;
    auto take = [&](size_t bytes) { char* p = ws + off; off += (bytes + 255) & ~(size_t)255; return p; };
    __bf16* kmat   = (__bf16*)take((size_t)BB * NN * DD * 2);   // 32 MB
    __bf16* vT     = (__bf16*)take((size_t)BB * DD * NN * 2);   // 32 MB
    __bf16* attn   = (__bf16*)take((size_t)BB * 16 * NN * 2);   //  8 MB
    __bf16* qmat   = (__bf16*)take((size_t)BB * 16 * DD * 2);
    float*  slotsA = (float*)take((size_t)BB * SS * DD * 4);
    float*  slotsB = (float*)take((size_t)BB * SS * DD * 4);
    float*  upd    = (float*)take((size_t)BB * SS * DD * 4);
    float*  rowsum = (float*)take((size_t)BB * SS * 4);
    (void)ws_size; (void)in_sizes; (void)n_in; (void)out_size;

    // one-time: slots init, attn pad rows, fused LN/K/V projection
    k_init_slots<<<(BB * SS * DD + 255) / 256, 256, 0, stream>>>(mu, lsig, noise, slotsA);
    {
        const size_t padElems = (size_t)BB * SS * NN;            // bf16 elements
        k_zero_attn_pad<<<(unsigned)(padElems / 8 / 256), 256, 0, stream>>>(attn);
    }
    k_lnkv<<<BB * NN / 128, 256, 0, stream>>>(inputs, g_in, beta_in, wk, bk, wv, bv, kmat, vT);

    float* cur = slotsA;
    float* nxt = slotsB;
    for (int it = 0; it < NITR; ++it) {
        k_qproj<<<BB, 64, 0, stream>>>(cur, g_sl, beta_sl, wq, bq, qmat, rowsum, upd);
        k_dots<<<BB * (NN / 128), 256, 0, stream>>>(qmat, kmat, attn, rowsum);
        k_updates<<<BB * 4, 128, 0, stream>>>(attn, vT, upd);
        float* outp = (it == NITR - 1) ? (float*)d_out : nxt;
        k_gru_mlp<<<BB * SS, 64, 0, stream>>>(upd, rowsum, cur, w_ih, b_ih, w_hh, b_hh,
                                              g_ff, beta_ff, w1, b1, w2, b2, outp);
        float* tmp = cur; cur = nxt; nxt = tmp;
    }
}
